// GroupedMLP_38757784879529
// MI455X (gfx1250) — matrix operand order repaired and checked by reading, not compiled, not run on hardware
//
#include <hip/hip_runtime.h>

#define E_ 8
#define H_ 1024
#define I_ 512
#define TM 64
#define SW 72      // weight panel stride (64 + 8 pad)

typedef __attribute__((ext_vector_type(16))) __bf16        v16bf;
typedef __attribute__((ext_vector_type(2)))  __bf16        v2bf;
typedef __attribute__((ext_vector_type(8)))  float         v8f;
typedef __attribute__((ext_vector_type(4)))  float         f32x4;
typedef __attribute__((ext_vector_type(4)))  unsigned int  u32x4;
typedef __attribute__((ext_vector_type(2)))  unsigned int  u32x2;

union FragU { u32x4 q[2]; v16bf v; };

__device__ __forceinline__ unsigned short f2bf(float f) {
  return __builtin_bit_cast(unsigned short, (__bf16)f);
}

// Two-source packed convert: lowers to a single v_cvt_pk_bf16_f32
__device__ __forceinline__ unsigned int pk2(float lo, float hi) {
  v2bf t;
  t.x = (__bf16)lo;
  t.y = (__bf16)hi;
  return __builtin_bit_cast(unsigned int, t);
}

__device__ __forceinline__ v8f vzero8() {
  v8f z;
#pragma unroll
  for (int i = 0; i < 8; ++i) z[i] = 0.0f;
  return z;
}

// A-matrix (16x32 bf16): lane r=lane&15 holds row r; low-half lanes K {0..7,16..23},
// high-half lanes K {8..15,24..31}; two b128 LDS reads.
__device__ __forceinline__ v16bf frag_a(const unsigned short* base, int stride,
                                        int row, int half, int k0) {
  FragU f;
  const unsigned short* p = base + row * stride + k0 + half * 8;
  f.q[0] = *(const u32x4*)(p);
  f.q[1] = *(const u32x4*)(p + 16);
  return f.v;
}

// B-matrix (32x16 bf16): lane n=lane&15 holds column n; low-half K {0..7,16..23},
// high-half K {8..15,24..31}, K-contiguous (weights transposed in LDS).
__device__ __forceinline__ v16bf frag_b(const unsigned short* base,
                                        int nrow, int half, int k0) {
  FragU f;
  const unsigned short* p = base + nrow * SW + k0 + half * 8;
  f.q[0] = *(const u32x4*)(p);
  f.q[1] = *(const u32x4*)(p + 16);
  return f.v;
}

__device__ __forceinline__ v8f wmma_bf16(v16bf a, v16bf b, v8f c) {
  return __builtin_amdgcn_wmma_f32_16x16x32_bf16(false, a, false, b,
                                                 (short)0, c, false, false);
}

// ---- panel staging: 256 cols x 64 K, fp32 global -> bf16 transposed LDS ----
__device__ __forceinline__ void p1_load(const float* __restrict__ Wgu, int e, int c,
                                        int kp, int tid, f32x4* R) {
#pragma unroll
  for (int i = 0; i < 4; ++i) {
    int u  = tid + 256 * i;
    int j4 = (u & 63) * 4;
    int k4 = (u >> 6) * 4;
    int gcol = c * 128 + j4 + ((j4 >> 7) * 384);  // j4>=128 -> up cols at 512+
    const float* g = Wgu + ((size_t)e * H_ + kp * 64 + k4) * (2 * I_) + gcol;
#pragma unroll
    for (int kk = 0; kk < 4; ++kk) R[i * 4 + kk] = *(const f32x4*)(g + kk * (2 * I_));
  }
}

__device__ __forceinline__ void p2_load(const float* __restrict__ Wdp, int e, int c2,
                                        int kp, int tid, f32x4* R) {
#pragma unroll
  for (int i = 0; i < 4; ++i) {
    int u  = tid + 256 * i;
    int j4 = (u & 63) * 4;
    int k4 = (u >> 6) * 4;
    const float* g = Wdp + ((size_t)e * I_ + kp * 64 + k4) * H_ + c2 * 256 + j4;
#pragma unroll
    for (int kk = 0; kk < 4; ++kk) R[i * 4 + kk] = *(const f32x4*)(g + kk * H_);
  }
}

// 4 k-values packed per column -> ds_store_b64
__device__ __forceinline__ void panel_store(unsigned short* buf, int tid, const f32x4* R) {
#pragma unroll
  for (int i = 0; i < 4; ++i) {
    int u  = tid + 256 * i;
    int j4 = (u & 63) * 4;
    int k4 = (u >> 6) * 4;
#pragma unroll
    for (int jj = 0; jj < 4; ++jj) {
      u32x2 p;
      p.x = pk2(R[i * 4 + 0][jj], R[i * 4 + 1][jj]);
      p.y = pk2(R[i * 4 + 2][jj], R[i * 4 + 3][jj]);
      *(u32x2*)(buf + (j4 + jj) * SW + k4) = p;
    }
  }
}

// LDS layout (ushort units):
//   Xs   @ 0      : 64 x 1032                     132096 B
//   W1a  @ 66048  : 256 x 72                       36864 B
//   W1b  @ 84480  : 256 x 72                       36864 B
//   Wd a/b @ 0 / 18432 : 256 x 72 each  (phase 2, alias Xs region)
//   Hs   @ 102912 : 64 x 520                       66560 B
// Total 272384 B (< 320 KB WGP LDS)
__global__ __launch_bounds__(256, 1)
void grouped_mlp_wmma(const float* __restrict__ X,
                      const float* __restrict__ Wgu,
                      const float* __restrict__ Wdp,
                      const int*   __restrict__ ntok,
                      float* __restrict__ out) {
  __shared__ __attribute__((aligned(16))) unsigned short smem[136192];
  unsigned short* Xs  = smem;
  unsigned short* W1a = smem + 66048;
  unsigned short* W1b = smem + 84480;
  unsigned short* Wda = smem;
  unsigned short* Wdb = smem + 18432;
  unsigned short* Hs  = smem + 102912;

  const int tid  = threadIdx.x;
  const int lane = tid & 31;
  const int wid  = tid >> 5;
  const int r16  = lane & 15;
  const int half = lane >> 4;
  const int mgrp = wid & 1;        // 2 M-groups, 2 m-tiles (32 rows) each
  const int ngrp = wid >> 1;       // 4 N-groups

  const int row0 = blockIdx.x * TM;

  int e = 0, cum = 0;
  while (e < E_ - 1 && row0 >= cum + ntok[e]) { cum += ntok[e]; ++e; }

  f32x4 R[16];
  p1_load(Wgu, e, 0, 0, tid, R);   // first weight panel in flight under X staging

  // ---- Stage X tile (TM x 1024) fp32 -> bf16; NT loads keep weights hot in L2 ----
  for (int l = tid; l < TM * 256; l += 256) {
    int rr = l >> 8;
    int cc = (l & 255) * 4;
    f32x4 f = __builtin_nontemporal_load((const f32x4*)(X + (size_t)(row0 + rr) * H_ + cc));
    u32x2 p;
    p.x = pk2(f.x, f.y);
    p.y = pk2(f.z, f.w);
    *(u32x2*)(Xs + rr * 1032 + cc) = p;
  }

  // ============ Phase 1: Hs = silu(X*Wg) .* (X*Wu) ============
  // 16 K-panels of 64, double-buffered; prefetch chain crosses chunk/phase bounds.
  for (int c = 0; c < 4; ++c) {
    v8f ag[2][2], au[2][2];
#pragma unroll
    for (int m = 0; m < 2; ++m)
#pragma unroll
      for (int n = 0; n < 2; ++n) { ag[m][n] = vzero8(); au[m][n] = vzero8(); }

    panel_store(W1a, tid, R);      // R holds this chunk's panel 0
    __syncthreads();

    for (int kp = 0; kp < 16; ++kp) {
      unsigned short* cur = (kp & 1) ? W1b : W1a;
      unsigned short* nxt = (kp & 1) ? W1a : W1b;
      if (kp + 1 < 16)      p1_load(Wgu, e, c, kp + 1, tid, R);
      else if (c + 1 < 4)   p1_load(Wgu, e, c + 1, 0, tid, R);
      else                  p2_load(Wdp, e, 0, 0, tid, R);
#pragma unroll
      for (int ks = 0; ks < 2; ++ks) {
        const int k0 = ks * 32;
        v16bf a0  = frag_a(Xs, 1032, mgrp * 32 +      r16, half, kp * 64 + k0);
        v16bf a1  = frag_a(Xs, 1032, mgrp * 32 + 16 + r16, half, kp * 64 + k0);
        v16bf bg0 = frag_b(cur,       ngrp * 32 +      r16, half, k0);
        v16bf bg1 = frag_b(cur,       ngrp * 32 + 16 + r16, half, k0);
        v16bf bu0 = frag_b(cur, 128 + ngrp * 32 +      r16, half, k0);
        v16bf bu1 = frag_b(cur, 128 + ngrp * 32 + 16 + r16, half, k0);
        ag[0][0] = wmma_bf16(a0, bg0, ag[0][0]);
        ag[0][1] = wmma_bf16(a0, bg1, ag[0][1]);
        ag[1][0] = wmma_bf16(a1, bg0, ag[1][0]);
        ag[1][1] = wmma_bf16(a1, bg1, ag[1][1]);
        au[0][0] = wmma_bf16(a0, bu0, au[0][0]);
        au[0][1] = wmma_bf16(a0, bu1, au[0][1]);
        au[1][0] = wmma_bf16(a1, bu0, au[1][0]);
        au[1][1] = wmma_bf16(a1, bu1, au[1][1]);
      }
      if (kp + 1 < 16) { panel_store(nxt, tid, R); __syncthreads(); }
    }
    // SiLU epilogue from C-layout (M = v + 8*half, N = lane&15)
#pragma unroll
    for (int m = 0; m < 2; ++m)
#pragma unroll
      for (int n = 0; n < 2; ++n) {
        int ncol  = c * 128 + ngrp * 32 + n * 16 + r16;
        int rbase = mgrp * 32 + m * 16 + half * 8;
#pragma unroll
        for (int v = 0; v < 8; ++v) {
          float g = ag[m][n][v];
          float u = au[m][n][v];
          float h = (g / (1.0f + __expf(-g))) * u;
          Hs[(rbase + v) * 520 + ncol] = f2bf(h);
        }
      }
  }

  __syncthreads();   // all Xs reads + Hs writes complete before Wda overwrites Xs

  // ============ Phase 2: out = Hs * Wd ============
  for (int c2 = 0; c2 < 4; ++c2) {
    v8f acc[2][4];
#pragma unroll
    for (int m = 0; m < 2; ++m)
#pragma unroll
      for (int i = 0; i < 4; ++i) acc[m][i] = vzero8();

    panel_store(Wda, tid, R);      // R holds this chunk's panel 0
    __syncthreads();

    for (int kp = 0; kp < 8; ++kp) {
      unsigned short* cur = (kp & 1) ? Wdb : Wda;
      unsigned short* nxt = (kp & 1) ? Wda : Wdb;
      if (kp + 1 < 8)       p2_load(Wdp, e, c2, kp + 1, tid, R);
      else if (c2 + 1 < 4)  p2_load(Wdp, e, c2 + 1, 0, tid, R);
#pragma unroll
      for (int ks = 0; ks < 2; ++ks) {
        const int k0 = ks * 32;
        v16bf a0 = frag_a(Hs, 520, mgrp * 32 +      r16, half, kp * 64 + k0);
        v16bf a1 = frag_a(Hs, 520, mgrp * 32 + 16 + r16, half, kp * 64 + k0);
#pragma unroll
        for (int i = 0; i < 4; ++i) {
          v16bf b = frag_b(cur, ngrp * 64 + i * 16 + r16, half, k0);
          acc[0][i] = wmma_bf16(a0, b, acc[0][i]);
          acc[1][i] = wmma_bf16(a1, b, acc[1][i]);
        }
      }
      if (kp + 1 < 8) { panel_store(nxt, tid, R); __syncthreads(); }
    }
#pragma unroll
    for (int m = 0; m < 2; ++m)
#pragma unroll
      for (int i = 0; i < 4; ++i) {
        int ncol = c2 * 256 + ngrp * 64 + i * 16 + r16;
        int rowb = row0 + mgrp * 32 + m * 16 + half * 8;
#pragma unroll
        for (int v = 0; v < 8; ++v) {
          __builtin_nontemporal_store(acc[m][i][v],
                                      out + (size_t)(rowb + v) * H_ + ncol);
        }
      }
  }
}

extern "C" void kernel_launch(void* const* d_in, const int* in_sizes, int n_in,
                              void* d_out, int out_size, void* d_ws, size_t ws_size,
                              hipStream_t stream) {
  (void)n_in; (void)out_size; (void)d_ws; (void)ws_size;
  const float* X   = (const float*)d_in[0];   // hidden_states  (T, H) f32
  const float* Wgu = (const float*)d_in[1];   // gate_up_proj   (E, H, 2I) f32
  const float* Wdp = (const float*)d_in[2];   // down_proj      (E, I, H) f32
  const int*   nt  = (const int*)d_in[3];     // num_tokens_per_expert (E,) i32
  float* out = (float*)d_out;                 // (T, H) f32

  const int T_total = in_sizes[0] / H_;       // 131072
  const int blocks  = T_total / TM;           // 2048
  grouped_mlp_wmma<<<blocks, 256, 0, stream>>>(X, Wgu, Wdp, nt, out);
}
